// IPAttnProcessorDouble_20822001451710
// MI455X (gfx1250) — matrix operand order repaired and checked by reading, not compiled, not run on hardware
//
#include <hip/hip_runtime.h>

// ---------------- problem constants ----------------
#define B_    8
#define S_    4096
#define C_    640
#define CD_   768
#define L_    85
#define H_    8
#define D_    80          // head dim
#define TTXT  77          // text tokens
#define NTOK  4
#define M_    (B_ * S_)   // 32768 rows for the big GEMMs
#define SCALE_ 0.11180339887498949f  // 1/sqrt(80)

typedef __attribute__((ext_vector_type(8)))  float        v8f;
typedef __attribute__((ext_vector_type(16))) __bf16       v16bf;
typedef __attribute__((ext_vector_type(4)))  unsigned int v4u;
typedef __attribute__((ext_vector_type(4)))  unsigned int u32x4;
typedef __attribute__((ext_vector_type(4)))  int          i32x4;
typedef __attribute__((ext_vector_type(8)))  int          i32x8;

union Frag { v16bf v; v4u q[2]; };

#if defined(__has_builtin)
#if __has_builtin(__builtin_amdgcn_tensor_load_to_lds)
#define USE_TDM 1
#endif
#endif

static __device__ __forceinline__ unsigned short f32_to_bf16(float f) {
  unsigned int u = __float_as_uint(f);
  unsigned int r = u + 0x7FFFu + ((u >> 16) & 1u);   // RNE
  return (unsigned short)(r >> 16);
}
static __device__ __forceinline__ float bf16_to_f32(unsigned short h) {
  return __uint_as_float(((unsigned int)h) << 16);
}
// generic LDS pointer low 32 bits == workgroup-relative LDS byte offset
static __device__ __forceinline__ unsigned lds_off(const void* p) {
  return (unsigned)(unsigned long long)p;
}

#ifdef USE_TDM
// 2D tile DMA: global (row-major, stride_elems bf16 per row) -> LDS (packed)
// tile_d0 = contiguous elems per row (K), tile_d1 = rows
static __device__ __forceinline__ void tdm_load_2d(unsigned lds_addr,
                                                   const void* gtile,
                                                   int tile_d0, int tile_d1,
                                                   int stride_elems) {
  unsigned long long ga = (unsigned long long)gtile;
  u32x4 g0;
  g0[0] = 1u;                                            // count=1, user mode
  g0[1] = lds_addr;                                      // lds_addr[31:0]
  g0[2] = (unsigned)(ga & 0xFFFFFFFFu);                  // global_addr[31:0]
  g0[3] = (unsigned)((ga >> 32) & 0x1FFFFFFu) | (2u << 30); // addr[56:32]|type=2
  i32x8 g1;
  g1[0] = (1 << 16);                                     // data_size=1 (2B)
  g1[1] = (tile_d0 & 0xFFFF) << 16;                      // tensor_dim0 lo16
  g1[2] = (tile_d1 & 0xFFFF) << 16;                      // tensor_dim1 lo16
  g1[3] = (tile_d0 & 0xFFFF) << 16;                      // tile_dim0
  g1[4] = (tile_d1 & 0xFFFF);                            // tile_dim1 (dim2=0)
  g1[5] = stride_elems;                                  // dim0_stride lo32
  g1[6] = 0;
  g1[7] = 0;
  i32x4 z4 = {0, 0, 0, 0};
#if __clang_major__ >= 23
  i32x8 z8 = {0, 0, 0, 0, 0, 0, 0, 0};
  __builtin_amdgcn_tensor_load_to_lds(g0, g1, z4, z4, z8, 0);
#else
  __builtin_amdgcn_tensor_load_to_lds(g0, g1, z4, z4, 0);
#endif
}
#endif

// ---------------- elementwise f32 -> bf16 convert ----------------
__global__ void cvt_bf16_kernel(const float* __restrict__ src,
                                unsigned short* __restrict__ dst, int n) {
  int i = blockIdx.x * blockDim.x + threadIdx.x;
  if (i < n) dst[i] = f32_to_bf16(src[i]);
}

// ---------------- f32 (K,N) -> bf16 transposed (N,K) ----------------
__global__ void cvt_bf16_tr_kernel(const float* __restrict__ src,
                                   unsigned short* __restrict__ dst) {
  int idx = blockIdx.x * blockDim.x + threadIdx.x;   // over 640*640
  int k = idx % C_;
  int n = idx / C_;
  dst[(size_t)n * C_ + k] = f32_to_bf16(src[(size_t)k * C_ + n]);
}

// ---------------- small projections (text K/V, ip K/V, ips K) ----------------
__global__ void proj_small_kernel(const float* __restrict__ enc,
                                  const float* __restrict__ W,
                                  float* __restrict__ out,
                                  int rowsPerB, int rowOff) {
  int idx = blockIdx.x * blockDim.x + threadIdx.x;
  int total = B_ * rowsPerB * C_;
  if (idx >= total) return;
  int n = idx % C_;
  int r = (idx / C_) % rowsPerB;
  int b = idx / (C_ * rowsPerB);
  const float* e = enc + ((size_t)b * L_ + rowOff + r) * CD_;
  float s = 0.f;
#pragma unroll 4
  for (int k = 0; k < CD_; ++k) s = fmaf(e[k], W[(size_t)k * C_ + n], s);
  out[idx] = s;
}

// ---------------- bf16 WMMA GEMM: (M x 640) @ (640 x 640) ----------------
// A: row-major bf16 [M][640]; WT: bf16 [N=640][K=640] (pre-transposed weights)
// mode 0: store bf16 (Q projection)
// mode 1: out_f32 = 0.5*(acc + bo[n]) + res[m,n]
#define BM 128
#define BN 64
#define BK 32

__global__ __launch_bounds__(256)
void gemm_bf16_kernel(const unsigned short* __restrict__ A,
                      const unsigned short* __restrict__ WT,
                      int mode,
                      float* __restrict__ outF,
                      unsigned short* __restrict__ outB,
                      const float* __restrict__ bo,
                      const float* __restrict__ res) {
#ifdef USE_TDM
  __shared__ __align__(16) unsigned short As[2][BM * BK];  // 2 x 8 KB
  __shared__ __align__(16) unsigned short Bs[2][BN * BK];  // 2 x 4 KB, [n][k]
#else
  __shared__ __align__(16) unsigned short As[1][BM * BK];
  __shared__ __align__(16) unsigned short Bs[1][BN * BK];
#endif

  const int tid    = threadIdx.x;
  const int lane   = tid & 31;
  const int wave   = tid >> 5;      // 0..7
  const int waveM  = wave >> 1;     // 0..3 -> 32-row strip
  const int waveN  = wave & 1;      // 0..1 -> 32-col strip
  const int blockM = blockIdx.y * BM;
  const int blockN = blockIdx.x * BN;

  v8f acc[2][2];
#pragma unroll
  for (int i = 0; i < 2; ++i)
#pragma unroll
    for (int j = 0; j < 2; ++j)
#pragma unroll
      for (int r = 0; r < 8; ++r) acc[i][j][r] = 0.f;

  const int lrow  = lane & 15;
  const int khalf = lane >> 4;          // 0 | 1
  const int nIter = 640 / BK;           // 20

#ifdef USE_TDM
  // prologue: fill buffer 0 via Tensor Data Mover (wave 0 issues the DMA)
  if (wave == 0) {
    tdm_load_2d(lds_off(&As[0][0]), A  + (size_t)blockM * 640, BK, BM, 640);
    tdm_load_2d(lds_off(&Bs[0][0]), WT + (size_t)blockN * 640, BK, BN, 640);
  }
#endif

  for (int it = 0; it < nIter; ++it) {
#ifdef USE_TDM
    const int buf = it & 1;
    if (wave == 0) {
      if (it + 1 < nIter) {
        const int k0 = (it + 1) * BK;
        tdm_load_2d(lds_off(&As[buf ^ 1][0]),
                    A + (size_t)blockM * 640 + k0, BK, BM, 640);
        tdm_load_2d(lds_off(&Bs[buf ^ 1][0]),
                    WT + (size_t)blockN * 640 + k0, BK, BN, 640);
        __builtin_amdgcn_s_wait_tensorcnt(2);  // current buffer resident
      } else {
        __builtin_amdgcn_s_wait_tensorcnt(0);
      }
    }
    __syncthreads();
#else
    const int buf = 0;
    {  // cooperative-load fallback (round-1 path)
      const int k0    = it * BK;
      const int arow  = tid >> 1;
      const int ahalf = (tid & 1) * 16;
      const unsigned short* ag = A + (size_t)(blockM + arow) * 640 + k0 + ahalf;
      v4u a0 = ((const v4u*)ag)[0];
      v4u a1 = ((const v4u*)ag)[1];
      *((v4u*)&As[0][arow * BK + ahalf])     = a0;
      *((v4u*)&As[0][arow * BK + ahalf + 8]) = a1;
      const int bn  = tid >> 2;             // 0..63
      const int bkq = (tid & 3) * 8;        // 0..24
      const unsigned short* bg = WT + (size_t)(blockN + bn) * 640 + k0 + bkq;
      *((v4u*)&Bs[0][bn * BK + bkq]) = *((const v4u*)bg);
    }
    __syncthreads();
#endif

    // ISA 7.12.2 fragment layouts
    Frag af[2], bfr[2];
    const int r0 = waveM * 32;
    const int c0 = waveN * 32;
#pragma unroll
    for (int i = 0; i < 2; ++i) {
      const unsigned short* ap = &As[buf][(r0 + i * 16 + lrow) * BK + khalf * 8];
      af[i].q[0] = *((const v4u*)ap);
      af[i].q[1] = *((const v4u*)(ap + 16));
    }
#pragma unroll
    for (int j = 0; j < 2; ++j) {
      const unsigned short* bp = &Bs[buf][(c0 + j * 16 + lrow) * BK + khalf * 8];
      bfr[j].q[0] = *((const v4u*)bp);
      bfr[j].q[1] = *((const v4u*)(bp + 16));
    }

#pragma unroll
    for (int i = 0; i < 2; ++i)
#pragma unroll
      for (int j = 0; j < 2; ++j)
        acc[i][j] = __builtin_amdgcn_wmma_f32_16x16x32_bf16(
            false, af[i].v, false, bfr[j].v, (short)0, acc[i][j], false, false);

    __syncthreads();
  }

  // epilogue: lane t holds (M = r + 8*(t/16), N = t%16), r=0..7
  const int lcol = lane & 15;
  const int lrh  = (lane >> 4) * 8;
#pragma unroll
  for (int i = 0; i < 2; ++i) {
#pragma unroll
    for (int j = 0; j < 2; ++j) {
      const int gN  = blockN + waveN * 32 + j * 16 + lcol;
      const int gM0 = blockM + waveM * 32 + i * 16 + lrh;
      if (mode == 0) {
#pragma unroll
        for (int r = 0; r < 8; ++r)
          outB[(size_t)(gM0 + r) * 640 + gN] = f32_to_bf16(acc[i][j][r]);
      } else {
        const float bias = bo[gN];
#pragma unroll
        for (int r = 0; r < 8; ++r) {
          const size_t idx = (size_t)(gM0 + r) * 640 + gN;
          outF[idx] = 0.5f * (acc[i][j][r] + bias) + res[idx];
        }
      }
    }
  }
}

// ---------------- fused attention (text 77 keys + IP 4+4 keys) ----------------
__global__ __launch_bounds__(256)
void attn_kernel(const unsigned short* __restrict__ qb,
                 const float* __restrict__ kx,   const float* __restrict__ vx,
                 const float* __restrict__ ipk,  const float* __restrict__ ipv,
                 const float* __restrict__ ipsk, const int* __restrict__ mask,
                 unsigned short* __restrict__ attn) {
  __shared__ float Qs[8][D_];
  __shared__ float Ps[8][80];   // 77 used
  __shared__ float Pc[8][4];

  const int w    = threadIdx.x >> 5;
  const int lane = threadIdx.x & 31;
  const int rowid = blockIdx.x * 8 + w;          // [0, B*H*S)
  const int b = rowid >> 15;
  const int h = (rowid >> 12) & 7;
  const int s = rowid & 4095;

  const unsigned short* qr = qb + ((size_t)b * S_ + s) * C_ + h * D_;
#pragma unroll
  for (int c = 0; c < 3; ++c) {
    int dd = lane + 32 * c;
    if (dd < D_) Qs[w][dd] = bf16_to_f32(qr[dd]);
  }
  __syncthreads();

  // ---- text scores: lane handles keys {lane, lane+32, lane+64} ----
  float z[3] = {-1e30f, -1e30f, -1e30f};
#pragma unroll
  for (int c = 0; c < 3; ++c) {
    int j = lane + 32 * c;
    if (j < TTXT) {
      const float* kr = kx + ((size_t)b * TTXT + j) * C_ + h * D_;
      float a = 0.f;
#pragma unroll 8
      for (int dd = 0; dd < D_; ++dd) a = fmaf(Qs[w][dd], kr[dd], a);
      z[c] = a * SCALE_;
    }
  }
  float m = fmaxf(fmaxf(z[0], z[1]), z[2]);
  for (int off = 16; off >= 1; off >>= 1) m = fmaxf(m, __shfl_xor(m, off, 32));
  float e[3], lsum = 0.f;
#pragma unroll
  for (int c = 0; c < 3; ++c) {
    int j = lane + 32 * c;
    e[c] = (j < TTXT) ? __expf(z[c] - m) : 0.f;
    lsum += e[c];
  }
  for (int off = 16; off >= 1; off >>= 1) lsum += __shfl_xor(lsum, off, 32);
  const float inv = 1.f / lsum;
#pragma unroll
  for (int c = 0; c < 3; ++c) {
    int j = lane + 32 * c;
    if (j < TTXT) Ps[w][j] = e[c] * inv;
  }

  // ---- IP scores: lanes 0-3 -> ip_k, lanes 4-7 -> ips_k ----
  float zi = -1e30f;
  if (lane < 8) {
    const int jp = lane & 3;
    const float* base = (lane < 4) ? ipk : ipsk;
    const float* kr = base + ((size_t)b * NTOK + jp) * C_ + h * D_;
    float a = 0.f;
#pragma unroll 8
    for (int dd = 0; dd < D_; ++dd) a = fmaf(Qs[w][dd], kr[dd], a);
    zi = a * SCALE_;
  }
  float mg = zi;
  mg = fmaxf(mg, __shfl_xor(mg, 1, 32));
  mg = fmaxf(mg, __shfl_xor(mg, 2, 32));
  float eg = __expf(zi - mg);
  float sg = eg;
  sg += __shfl_xor(sg, 1, 32);
  sg += __shfl_xor(sg, 2, 32);
  float pg = eg / sg;
  float po = __shfl_xor(pg, 4, 32);
  if (lane < 4) {
    const float factor = mask[s] ? 0.01f : 1.0f;
    Pc[w][lane] = fmaxf(pg, po) * factor;   // (a+b+|a-b|)/2 == max(a,b)
  }
  __syncthreads();

  // ---- P @ V  (+ IP part, IP_SCALE = 1) ----
  unsigned short* orow = attn + ((size_t)b * S_ + s) * C_ + h * D_;
#pragma unroll
  for (int c = 0; c < 3; ++c) {
    int dd = lane + 32 * c;
    if (dd < D_) {
      float a = 0.f;
      const float* vcol = vx + (size_t)b * TTXT * C_ + h * D_ + dd;
#pragma unroll 7
      for (int j = 0; j < TTXT; ++j) a = fmaf(Ps[w][j], vcol[(size_t)j * C_], a);
      const float* ivcol = ipv + (size_t)b * NTOK * C_ + h * D_ + dd;
#pragma unroll
      for (int j = 0; j < NTOK; ++j) a = fmaf(Pc[w][j], ivcol[(size_t)j * C_], a);
      orow[dd] = f32_to_bf16(a);
    }
  }
}

// ---------------- host-side launch ----------------
extern "C" void kernel_launch(void* const* d_in, const int* in_sizes, int n_in,
                              void* d_out, int out_size, void* d_ws, size_t ws_size,
                              hipStream_t stream) {
  const float* hs    = (const float*)d_in[0];
  const float* enc   = (const float*)d_in[1];
  const int*   mask  = (const int*)  d_in[2];
  const float* Wq    = (const float*)d_in[3];
  const float* Wk    = (const float*)d_in[4];
  const float* Wv    = (const float*)d_in[5];
  const float* Wkip  = (const float*)d_in[6];
  const float* Wvip  = (const float*)d_in[7];
  const float* Wksip = (const float*)d_in[8];
  const float* Wo    = (const float*)d_in[9];
  const float* bo    = (const float*)d_in[10];
  float* out = (float*)d_out;

  char* ws = (char*)d_ws;
  size_t off = 0;
  auto carve = [&](size_t bytes) -> char* {
    char* p = ws + off;
    off = (off + bytes + 255) & ~(size_t)255;
    return p;
  };
  const size_t nHS = (size_t)M_ * C_;
  unsigned short* hsb    = (unsigned short*)carve(nHS * 2);
  unsigned short* qbf    = (unsigned short*)carve(nHS * 2);
  unsigned short* attnbf = (unsigned short*)carve(nHS * 2);
  unsigned short* wqt    = (unsigned short*)carve((size_t)C_ * C_ * 2); // [n][k]
  unsigned short* wot    = (unsigned short*)carve((size_t)C_ * C_ * 2); // [n][k]
  float* kx   = (float*)carve((size_t)B_ * TTXT * C_ * 4);
  float* vx   = (float*)carve((size_t)B_ * TTXT * C_ * 4);
  float* ipkv = (float*)carve((size_t)B_ * NTOK * C_ * 4);
  float* ipvv = (float*)carve((size_t)B_ * NTOK * C_ * 4);
  float* ipsk = (float*)carve((size_t)B_ * NTOK * C_ * 4);

  // 1) convert activations; transpose+convert GEMM weights to bf16 [n][k]
  cvt_bf16_kernel<<<(int)(nHS / 256), 256, 0, stream>>>(hs, hsb, (int)nHS);
  cvt_bf16_tr_kernel<<<(C_ * C_) / 256, 256, 0, stream>>>(Wq, wqt);
  cvt_bf16_tr_kernel<<<(C_ * C_) / 256, 256, 0, stream>>>(Wo, wot);

  // 2) small K/V projections
  {
    const int nT = B_ * TTXT * C_;
    const int nI = B_ * NTOK * C_;
    proj_small_kernel<<<nT / 256, 256, 0, stream>>>(enc, Wk,    kx,   TTXT, 0);
    proj_small_kernel<<<nT / 256, 256, 0, stream>>>(enc, Wv,    vx,   TTXT, 0);
    proj_small_kernel<<<nI / 256, 256, 0, stream>>>(enc, Wkip,  ipkv, NTOK, TTXT);
    proj_small_kernel<<<nI / 256, 256, 0, stream>>>(enc, Wvip,  ipvv, NTOK, TTXT);
    proj_small_kernel<<<nI / 256, 256, 0, stream>>>(enc, Wksip, ipsk, NTOK, TTXT + NTOK);
  }

  // 3) Q projection: TDM-fed bf16 WMMA GEMM, bf16 out
  {
    dim3 grid(C_ / BN, M_ / BM);   // (10, 256)
    gemm_bf16_kernel<<<grid, 256, 0, stream>>>(hsb, wqt, 0, nullptr, qbf,
                                               nullptr, nullptr);
  }

  // 4) fused attention
  attn_kernel<<<(B_ * H_ * S_) / 8, 256, 0, stream>>>(qbf, kx, vx, ipkv, ipvv,
                                                      ipsk, mask, attnbf);

  // 5) output projection, fused epilogue: 0.5*(X@Wo + bo) + residual
  {
    dim3 grid(C_ / BN, M_ / BM);
    gemm_bf16_kernel<<<grid, 256, 0, stream>>>(attnbf, wot, 1, out, nullptr,
                                               bo, hs);
  }
}
